// DecoderAttention_55061480735339
// MI455X (gfx1250) — matrix operand order repaired and checked by reading, not compiled, not run on hardware
//
#include <hip/hip_runtime.h>

// ---------------------------------------------------------------------------
// DecoderAttention for MI455X (gfx1250, wave32).
// All heavy math through v_wmma_f32_16x16x32_bf16 (f32 accumulate).
// GEMM tiles staged into LDS with GLOBAL_LOAD_ASYNC_TO_LDS_B128 (ASYNCcnt)
// when the toolchain exposes the builtin; sync fallback otherwise.
// Shapes: V=32000 H=512 B=16 T=128. All dims divide WMMA/block tiles exactly.
// ---------------------------------------------------------------------------

typedef __attribute__((ext_vector_type(16))) __bf16 v16bf;
typedef __attribute__((ext_vector_type(8)))  __bf16 v8bf;
typedef __attribute__((ext_vector_type(8)))  float  v8f;

#define DEV __device__ __forceinline__

static constexpr int Bn   = 16;
static constexpr int Tn   = 128;
static constexpr int Hn   = 512;
static constexpr int Vn   = 32000;
static constexpr int G4H  = 2048;    // 4*H
static constexpr int K2H  = 1024;    // 2*H
static constexpr int Mrow = Bn * Tn; // 2048 flattened (b,t) rows

#if __has_builtin(__builtin_amdgcn_global_load_async_to_lds_b128) && \
    __has_builtin(__builtin_amdgcn_s_wait_asynccnt)
#define HAS_ASYNC_LDS 1
#else
#define HAS_ASYNC_LDS 0
#endif

#if HAS_ASYNC_LDS
// Builtin signature (from clang diagnostic): first arg is
// 'vector_size(16) int __device__*' (AS1), second the AS3 twin.
typedef int v4i_ __attribute__((vector_size(16)));
typedef __attribute__((address_space(1))) v4i_* g_v4ip;
typedef __attribute__((address_space(3))) v4i_* l_v4ip;
// 16-byte global -> LDS async copy. Low 32 bits of a generic shared pointer
// are the LDS byte offset; global generic
// pointers pass through unchanged.
DEV void async_cp16(const __bf16* gsrc, __bf16* ldst) {
  g_v4ip g = (g_v4ip)(size_t)gsrc;
  l_v4ip l = (l_v4ip)(unsigned int)(size_t)ldst;
  __builtin_amdgcn_global_load_async_to_lds_b128(g, l, 0, 0);
}
DEV void wait_async0() { __builtin_amdgcn_s_wait_asynccnt(0); }
#else
DEV void wait_async0() {}
#endif

DEV v8f v8f_zero() { v8f z = {0.f,0.f,0.f,0.f,0.f,0.f,0.f,0.f}; return z; }

DEV v8f wmma_bf16(v16bf a, v16bf b, v8f c) {
  // (neg_a, A, neg_b, B, c_mod, C, reuse_a, reuse_b)
  return __builtin_amdgcn_wmma_f32_16x16x32_bf16(false, a, false, b, (short)0, c, false, false);
}

DEV v16bf make_frag(v8bf lo, v8bf hi) {
  v16bf r;
#pragma unroll
  for (int i = 0; i < 8; ++i) { r[i] = lo[i]; r[i + 8] = hi[i]; }
  return r;
}

// A fragment (16x32, MxK) from row-major [.., ld] source.
// Lane l: row = l&15 ; K = (l>>4)*16 + {0..15}.
DEV v16bf load_a_frag(const __bf16* base, int ld) {
  int lane = threadIdx.x & 31;
  int m  = lane & 15;
  int kb = (lane >> 4) * 16;
  v8bf lo = *(const v8bf*)(base + m * ld + kb);
  v8bf hi = *(const v8bf*)(base + m * ld + kb + 8);
  return make_frag(lo, hi);
}

// B fragment (32x16, KxN) where the source is W[N][K] row-major (B[k][n]=W[n][k]).
// Lane l: col n = l&15 ; K = (l>>4)*16 + {0..15} (contiguous in W's row).
DEV v16bf load_b_frag(const __bf16* base, int ld) {
  int lane = threadIdx.x & 31;
  int n  = lane & 15;
  int kb = (lane >> 4) * 16;
  v8bf lo = *(const v8bf*)(base + n * ld + kb);
  v8bf hi = *(const v8bf*)(base + n * ld + kb + 8);
  return make_frag(lo, hi);
}

DEV float sigmoidf_(float x) { return 1.0f / (1.0f + __expf(-x)); }

// Grid-wide barrier over NWG resident blocks (atomic counter, monotone target).
__device__ void grid_sync(unsigned* cnt, unsigned target) {
  __syncthreads();
  if (threadIdx.x == 0) {
    __hip_atomic_fetch_add(cnt, 1u, __ATOMIC_ACQ_REL, __HIP_MEMORY_SCOPE_AGENT);
    while (__hip_atomic_load(cnt, __ATOMIC_ACQUIRE, __HIP_MEMORY_SCOPE_AGENT) < target) {
      __builtin_amdgcn_s_sleep(1);
    }
  }
  __syncthreads();
}

// ------------------------------ small kernels ------------------------------

__global__ void k_cvt_bf16(const float* __restrict__ src, __bf16* __restrict__ dst,
                           int nquads) {
  int i = blockIdx.x * blockDim.x + threadIdx.x;
  if (i >= nquads) return;
  float4 v = *(const float4*)(src + 4 * (size_t)i);
  __bf16* o = dst + 4 * (size_t)i;
  o[0] = (__bf16)v.x; o[1] = (__bf16)v.y; o[2] = (__bf16)v.z; o[3] = (__bf16)v.w;
}

__global__ void k_embed(const int* __restrict__ idxs, const float* __restrict__ emb,
                        __bf16* __restrict__ x_bf) {
  int i = blockIdx.x * blockDim.x + threadIdx.x;        // over Mrow * (H/4)
  if (i >= Mrow * (Hn / 4)) return;
  int row = i >> 7;            // H/4 = 128 quads per row
  int hq  = (i & 127) * 4;
  int r = idxs[row];
  float4 v = *(const float4*)&emb[(size_t)r * Hn + hq];
  __bf16* o = x_bf + (size_t)row * Hn + hq;
  o[0] = (__bf16)v.x; o[1] = (__bf16)v.y; o[2] = (__bf16)v.z; o[3] = (__bf16)v.w;
}

__global__ void k_zero_u32(unsigned* p) {
  if (threadIdx.x == 0 && blockIdx.x == 0) *p = 0u;
}

// ------------------------- LDS-tiled WMMA GEMM -----------------------------
// C[M,N] = A[M,K] @ W[N,K]^T + bias[N].  Block tile 128x128, K-chunk 32,
// double-buffered LDS with async global->LDS staging overlapped with WMMA.
// 8 waves in a 2(M)x4(N) grid; each wave owns a 64x32 sub-tile (4x2 frags).

__global__ __launch_bounds__(256)
void k_gemm_nt(const __bf16* __restrict__ A, const __bf16* __restrict__ W,
               const float* __restrict__ bias, float* __restrict__ C,
               int N, int K) {
  __shared__ __align__(32) __bf16 As[2][128][32];
  __shared__ __align__(32) __bf16 Bs[2][128][32];
  const int n0 = blockIdx.x * 128, m0 = blockIdx.y * 128;
  const int tid = threadIdx.x, wave = tid >> 5, lane = tid & 31;
  const int wm = wave >> 2;   // 0..1 -> M offset wm*64
  const int wn = wave & 3;    // 0..3 -> N offset wn*32
  const int r0 = tid >> 2, q0 = (tid & 3) * 8;   // this thread's two 16B chunks

  v8f acc[4][2];
#pragma unroll
  for (int f = 0; f < 4; ++f)
#pragma unroll
    for (int g = 0; g < 2; ++g) acc[f][g] = v8f_zero();

  auto stage = [&](int kk, int buf) {
#if HAS_ASYNC_LDS
    async_cp16(&A[(size_t)(m0 + r0) * K + kk + q0],      &As[buf][r0][q0]);
    async_cp16(&W[(size_t)(n0 + r0) * K + kk + q0],      &Bs[buf][r0][q0]);
    async_cp16(&A[(size_t)(m0 + r0 + 64) * K + kk + q0], &As[buf][r0 + 64][q0]);
    async_cp16(&W[(size_t)(n0 + r0 + 64) * K + kk + q0], &Bs[buf][r0 + 64][q0]);
#else
    *(v8bf*)&As[buf][r0][q0]      = *(const v8bf*)&A[(size_t)(m0 + r0) * K + kk + q0];
    *(v8bf*)&Bs[buf][r0][q0]      = *(const v8bf*)&W[(size_t)(n0 + r0) * K + kk + q0];
    *(v8bf*)&As[buf][r0 + 64][q0] = *(const v8bf*)&A[(size_t)(m0 + r0 + 64) * K + kk + q0];
    *(v8bf*)&Bs[buf][r0 + 64][q0] = *(const v8bf*)&W[(size_t)(n0 + r0 + 64) * K + kk + q0];
#endif
  };

  stage(0, 0);
  const int nk = K / 32;
  for (int kc = 0; kc < nk; ++kc) {
    const int buf = kc & 1;
    wait_async0();            // tile[buf] landed in LDS (this wave's copies)
    __syncthreads();          // all waves' copies visible; prev reads of buf^1 done
    if (kc + 1 < nk) stage((kc + 1) * 32, buf ^ 1);   // overlap with compute

    v16bf af[4], bfr[2];
#pragma unroll
    for (int f = 0; f < 4; ++f) af[f]  = load_a_frag(&As[buf][wm * 64 + f * 16][0], 32);
#pragma unroll
    for (int g = 0; g < 2; ++g) bfr[g] = load_b_frag(&Bs[buf][wn * 32 + g * 16][0], 32);
#pragma unroll
    for (int f = 0; f < 4; ++f)
#pragma unroll
      for (int g = 0; g < 2; ++g) acc[f][g] = wmma_bf16(af[f], bfr[g], acc[f][g]);
  }

  const int rr = (lane >> 4) * 8, nn = lane & 15;
#pragma unroll
  for (int f = 0; f < 4; ++f)
#pragma unroll
    for (int g = 0; g < 2; ++g) {
      int nb = n0 + wn * 32 + g * 16 + nn;
      float bv = bias[nb];
#pragma unroll
      for (int r = 0; r < 8; ++r) {
        int m = m0 + wm * 64 + f * 16 + rr + r;
        C[(size_t)m * N + nb] = acc[f][g][r] + bv;
      }
    }
}

// ------------------------------- LSTM --------------------------------------
// 64 persistent WGs x 128 threads. WG g owns hidden slice j0=[g*8, g*8+8):
// 32 rows of W_hh (4 gates x 8) cached in LDS as bf16 (32 KB). WMMA with
// M=batch(16), N=32 slice rows, K=512. h ping-pongs in global bf16; one
// grid barrier per timestep.

#define LSTM_WGS 64

__global__ __launch_bounds__(128)
void k_lstm(const float* __restrict__ xz,      // [B,T,4H] f32 (has b_ih)
            const __bf16* __restrict__ Whh_bf, // [4H,H] bf16
            const float* __restrict__ b_hh,    // [4H]
            const float* __restrict__ h0,      // [B,H]
            const float* __restrict__ c0,      // [B,H]
            __bf16* __restrict__ hglob,        // [2][B*H] ping-pong
            __bf16* __restrict__ dec_bf,       // [B*T, H]
            unsigned* __restrict__ sync) {
  __shared__ __align__(32) __bf16 Wsl[32][Hn];   // 32 KB W_hh slice
  __shared__ float zbuf[Bn][32];                 // gate pre-activations
  __shared__ float cbuf[Bn][8];                  // cell-state slice

  const int g = blockIdx.x, j0 = g * 8;
  const int tid = threadIdx.x, wave = tid >> 5, lane = tid & 31;

  // Stage W_hh slice into LDS (rows lr = q*8+jj  ->  global row q*512+j0+jj).
  for (int c = tid; c < 32 * (Hn / 8); c += 128) {
    int lr = c >> 6, col = (c & 63) * 8;
    int q = lr >> 3, jj = lr & 7;
    const __bf16* src = &Whh_bf[(size_t)(q * Hn + j0 + jj) * Hn + col];
#if HAS_ASYNC_LDS
    async_cp16(src, &Wsl[lr][col]);
#else
    *(v8bf*)&Wsl[lr][col] = *(const v8bf*)src;
#endif
  }
  wait_async0();
  { // init c and h (this WG's slice; 64 WGs cover all of H)
    int bb = tid >> 3, jj = tid & 7;
    cbuf[bb][jj] = c0[bb * Hn + j0 + jj];
    hglob[bb * Hn + j0 + jj] = (__bf16)h0[bb * Hn + j0 + jj];
  }
  __threadfence();
  grid_sync(sync, LSTM_WGS * 1u);

  for (int t = 0; t < Tn; ++t) {
    const __bf16* hin  = hglob + (t & 1) * (Bn * Hn);
    __bf16*       hout = hglob + ((t + 1) & 1) * (Bn * Hn);

    if (wave < 2) {                 // each wave: one 16-col N-tile
      v8f acc = v8f_zero();
#pragma unroll 4
      for (int kc = 0; kc < Hn / 32; ++kc) {
        v16bf a = load_a_frag(hin + kc * 32, Hn);
        v16bf b = load_b_frag(&Wsl[wave * 16][kc * 32], Hn);
        acc = wmma_bf16(a, b, acc);
      }
      int lr = wave * 16 + (lane & 15);
      int q = lr >> 3, jj = lr & 7;
      int grow = q * Hn + j0 + jj;
      float bh = b_hh[grow];
#pragma unroll
      for (int r = 0; r < 8; ++r) {
        int bb = (lane >> 4) * 8 + r;
        zbuf[bb][lr] = acc[r] + xz[(size_t)(bb * Tn + t) * G4H + grow] + bh;
      }
    }
    __syncthreads();
    { // element-wise gates: 128 (b,jj) pairs, 1 per thread
      int bb = tid >> 3, jj = tid & 7;
      float zi = zbuf[bb][jj],      zf = zbuf[bb][8 + jj];
      float zg = zbuf[bb][16 + jj], zo = zbuf[bb][24 + jj];
      float cc = sigmoidf_(zf) * cbuf[bb][jj] + sigmoidf_(zi) * tanhf(zg);
      cbuf[bb][jj] = cc;
      __bf16 hb = (__bf16)(sigmoidf_(zo) * tanhf(cc));
      hout[bb * Hn + j0 + jj] = hb;
      dec_bf[(size_t)(bb * Tn + t) * Hn + j0 + jj] = hb;
    }
    __threadfence();
    grid_sync(sync, LSTM_WGS * (unsigned)(t + 2));
  }
}

// ------------------------------ Attention ----------------------------------
// One WG per (batch, 16-row t-tile): scores = dec@enc^T (WMMA), row softmax,
// ctx = dist@enc (WMMA, enc transposed 64 cols at a time through LDS).
// Writes cat = [dec | ctx] in bf16.

__global__ __launch_bounds__(128)
void k_attn(const __bf16* __restrict__ dec_bf,   // [B*T, H]
            const __bf16* __restrict__ enc_bf,   // [B, T, H]
            __bf16* __restrict__ cat_bf) {       // [B*T, 2H]
  __shared__ float  sc[16][Tn];                  // 8 KB scores
  __shared__ __bf16 distb[16][Tn];               // 4 KB softmaxed dist
  __shared__ __align__(32) __bf16 encT[64][136]; // 17 KB transposed enc chunk

  const int b = blockIdx.x >> 3, tt = blockIdx.x & 7, t0 = tt * 16;
  const int tid = threadIdx.x, wave = tid >> 5, lane = tid & 31;
  const __bf16* decA = dec_bf + (size_t)(b * Tn + t0) * Hn;   // [16,H]
  const __bf16* encB = enc_bf + (size_t)b * Tn * Hn;          // [T,H]

  // 1) scores[16 x 128]: A = dec rows, B = enc rows (already W[N,K] form).
  for (int nt = wave; nt < 8; nt += 4) {
    v8f acc = v8f_zero();
#pragma unroll 4
    for (int kc = 0; kc < Hn / 32; ++kc) {
      v16bf a  = load_a_frag(decA + kc * 32, Hn);
      v16bf bb = load_b_frag(encB + (size_t)(nt * 16) * Hn + kc * 32, Hn);
      acc = wmma_bf16(a, bb, acc);
    }
#pragma unroll
    for (int r = 0; r < 8; ++r)
      sc[(lane >> 4) * 8 + r][nt * 16 + (lane & 15)] = acc[r];
  }
  __syncthreads();

  // 2) softmax over s (one thread per row), and copy dec into cat[:, :H].
  if (tid < 16) {
    float mx = -3.0e38f;
    for (int s = 0; s < Tn; ++s) mx = fmaxf(mx, sc[tid][s]);
    float sum = 0.f;
    for (int s = 0; s < Tn; ++s) { float e = __expf(sc[tid][s] - mx); sc[tid][s] = e; sum += e; }
    float inv = 1.0f / sum;
    for (int s = 0; s < Tn; ++s) distb[tid][s] = (__bf16)(sc[tid][s] * inv);
  }
  for (int idx = tid; idx < 16 * Hn; idx += 128) {
    int t = idx >> 9, h = idx & (Hn - 1);
    cat_bf[(size_t)(b * Tn + t0 + t) * K2H + h] = decA[(size_t)t * Hn + h];
  }
  __syncthreads();

  // 3) ctx[16 x 512] = dist[16x128] @ enc[128x512]; K = s = 128.
  v16bf afr[4];
#pragma unroll
  for (int kc = 0; kc < 4; ++kc) afr[kc] = load_a_frag(&distb[0][kc * 32], Tn);

  for (int hc = 0; hc < Hn / 64; ++hc) {
    __syncthreads();
    for (int idx = tid; idx < Tn * 64; idx += 128) {   // encT[hh][s] = enc[s][h0+hh]
      int s = idx >> 6, hh = idx & 63;
      encT[hh][s] = encB[(size_t)s * Hn + hc * 64 + hh];
    }
    __syncthreads();
    v8f acc = v8f_zero();
#pragma unroll
    for (int kc = 0; kc < 4; ++kc)
      acc = wmma_bf16(afr[kc], load_b_frag(&encT[wave * 16][kc * 32], 136), acc);
    const int nn = lane & 15;
#pragma unroll
    for (int r = 0; r < 8; ++r) {
      int m = t0 + (lane >> 4) * 8 + r;
      cat_bf[(size_t)(b * Tn + m) * K2H + Hn + hc * 64 + wave * 16 + nn] = (__bf16)acc[r];
    }
  }
}

// ------------------------------- driver ------------------------------------

extern "C" void kernel_launch(void* const* d_in, const int* in_sizes, int n_in,
                              void* d_out, int out_size, void* d_ws, size_t ws_size,
                              hipStream_t stream) {
  (void)in_sizes; (void)n_in; (void)out_size; (void)ws_size;
  const int*   indices = (const int*)  d_in[0];
  const float* enc     = (const float*)d_in[1];
  const float* h0      = (const float*)d_in[2];
  const float* c0      = (const float*)d_in[3];
  const float* emb     = (const float*)d_in[4];
  const float* W_ih    = (const float*)d_in[5];
  const float* W_hh    = (const float*)d_in[6];
  const float* b_ih    = (const float*)d_in[7];
  const float* b_hh    = (const float*)d_in[8];
  const float* W_out   = (const float*)d_in[9];
  const float* b_out   = (const float*)d_in[10];
  float* out = (float*)d_out;

  // workspace carve (~97 MB)
  char* w = (char*)d_ws;
  auto take = [&](size_t bytes) -> char* {
    char* p = w; w += (bytes + 255) & ~(size_t)255; return p;
  };
  __bf16* wih_bf  = (__bf16*)take((size_t)G4H * Hn * 2);
  __bf16* whh_bf  = (__bf16*)take((size_t)G4H * Hn * 2);
  __bf16* wout_bf = (__bf16*)take((size_t)Vn  * K2H * 2);
  __bf16* enc_bf  = (__bf16*)take((size_t)Bn * Tn * Hn * 2);
  __bf16* x_bf    = (__bf16*)take((size_t)Mrow * Hn * 2);
  float*  xz      = (float*) take((size_t)Mrow * G4H * 4);
  __bf16* hglob   = (__bf16*)take((size_t)2 * Bn * Hn * 2);
  __bf16* dec_bf  = (__bf16*)take((size_t)Mrow * Hn * 2);
  __bf16* cat_bf  = (__bf16*)take((size_t)Mrow * K2H * 2);
  unsigned* sync  = (unsigned*)take(256);

  const int CT = 256;
  // f32 -> bf16 staging
  k_cvt_bf16<<<(G4H * Hn / 4 + CT - 1) / CT, CT, 0, stream>>>(W_ih, wih_bf, G4H * Hn / 4);
  k_cvt_bf16<<<(G4H * Hn / 4 + CT - 1) / CT, CT, 0, stream>>>(W_hh, whh_bf, G4H * Hn / 4);
  k_cvt_bf16<<<(Vn * K2H / 4 + CT - 1) / CT, CT, 0, stream>>>(W_out, wout_bf, Vn * K2H / 4);
  k_cvt_bf16<<<(Bn * Tn * Hn / 4 + CT - 1) / CT, CT, 0, stream>>>(enc, enc_bf, Bn * Tn * Hn / 4);
  // x = emb[indices]
  k_embed<<<(Mrow * (Hn / 4) + CT - 1) / CT, CT, 0, stream>>>(indices, emb, x_bf);
  // xz = x @ W_ih^T + b_ih      (M=2048, N=2048, K=512)
  k_gemm_nt<<<dim3(G4H / 128, Mrow / 128), 256, 0, stream>>>(x_bf, wih_bf, b_ih, xz, G4H, Hn);
  // LSTM recurrence (persistent, grid-synced)
  k_zero_u32<<<1, 32, 0, stream>>>(sync);
  k_lstm<<<LSTM_WGS, 128, 0, stream>>>(xz, whh_bf, b_hh, h0, c0, hglob, dec_bf, sync);
  // attention + cat
  k_attn<<<Bn * 8, 128, 0, stream>>>(dec_bf, enc_bf, cat_bf);
  // out = cat @ W_out^T + b_out (M=2048, N=32000, K=1024) — dominant kernel
  k_gemm_nt<<<dim3(Vn / 128, Mrow / 128), 256, 0, stream>>>(cat_bf, wout_bf, b_out, out, Vn, K2H);
}
